// STML_GCL_loss_47897475285637
// MI455X (gfx1250) — compile-verified, matrix operand order checked by reading, not run on hardware
//
#include <hip/hip_runtime.h>
#include <hip/hip_bf16.h>
#include <math.h>

// ---------------- CDNA5 WMMA / TDM types ----------------
typedef __attribute__((ext_vector_type(16))) __bf16 v16bf;
typedef __attribute__((ext_vector_type(8)))  float  v8f;
typedef __attribute__((ext_vector_type(4)))  unsigned int v4u;
typedef __attribute__((ext_vector_type(8)))  int v8i;
typedef __attribute__((ext_vector_type(4)))  int v4i;

union FragBF { v16bf v; uint4 q[2]; };

#if defined(__has_builtin)
#  if __has_builtin(__builtin_amdgcn_tensor_load_to_lds)
#    define HAS_TDM 1
#  endif
#endif
#ifndef HAS_TDM
#  define HAS_TDM 0
#endif

__device__ __forceinline__ unsigned short f2bf(float x) {
    union { float f; unsigned u; } v; v.f = x;
    unsigned r = v.u + 0x7FFFu + ((v.u >> 16) & 1u);   // round-to-nearest-even
    return (unsigned short)(r >> 16);
}

__device__ __forceinline__ unsigned lds_off(const void* p) {
    return (unsigned)(unsigned long long)(size_t)p;    // low 32 bits = LDS byte addr
}

#if HAS_TDM
// Issue one TDM 2D-tile load: global (rows x cols bf16, row stride = strideElems)
// -> LDS at ldsOff, packed contiguously (tile_dim0 = cols). D# per ISA ch.8.
__device__ __forceinline__ void tdm_load_tile(
    const unsigned short* gbase, unsigned ldsOff,
    int rows, int cols, long long strideElems,
    long long tensorD0, long long tensorD1)
{
    unsigned long long ga = (unsigned long long)(size_t)gbase;
    v4u g0;
    g0[0] = 1u;                                              // count=1 (user D#)
    g0[1] = ldsOff;                                          // lds_addr
    g0[2] = (unsigned)(ga & 0xFFFFFFFFull);                  // global_addr[95:64]
    g0[3] = (unsigned)((ga >> 32) & 0x01FFFFFFull) | (2u << 30); // addr msbs | type=2
    unsigned td0 = (unsigned)tensorD0, td1 = (unsigned)tensorD1;
    unsigned long long s0 = (unsigned long long)strideElems;
    v8i g1;
    g1[0] = (int)(1u << 16);                                 // mask=0, data_size=1 (2B)
    g1[1] = (int)((td0 & 0xFFFFu) << 16);                    // tensor_dim0[15:0]
    g1[2] = (int)((td0 >> 16) | ((td1 & 0xFFFFu) << 16));    // td0[31:16] | td1[15:0]
    g1[3] = (int)((td1 >> 16) | ((unsigned)cols << 16));     // td1[31:16] | tile_dim0
    g1[4] = (int)((unsigned)rows & 0xFFFFu);                 // tile_dim1, tile_dim2=0
    g1[5] = (int)(s0 & 0xFFFFFFFFull);                       // dim0_stride[31:0]
    g1[6] = (int)((s0 >> 32) & 0xFFFFull);                   // dim0_stride[47:32]
    g1[7] = 0;
    v4i z4 = {0, 0, 0, 0};
#if __clang_major__ >= 23
    v8i z8 = {0, 0, 0, 0, 0, 0, 0, 0};
    __builtin_amdgcn_tensor_load_to_lds(g0, g1, z4, z4, z8, 0);
#else
    __builtin_amdgcn_tensor_load_to_lds(g0, g1, z4, z4, 0);
#endif
}
#endif

// ---------------------------------------------------------------------------
// Tiled WMMA GEMM:  C[Mp x Np] = A[Mp x K] * B[Np x K]^T   (bf16 in, f32 out)
// 256 threads = 8 waves; block tile 64x64; each wave owns a 16x32 strip
// (A-frag reused across two WMMAs). A/B tiles staged in LDS; on CDNA5 the
// Tensor Data Mover DMAs the tiles (double-buffered, overlapped with compute).
// Mp, Np, K must be multiples of 64.
// ---------------------------------------------------------------------------
#define BM 64
#define BN 64
#define BK 64

__global__ __launch_bounds__(256) void wmma_gemm_nt_tiled(
    const unsigned short* __restrict__ A,
    const unsigned short* __restrict__ B,
    float* __restrict__ C,
    int Mp, int Np, int K)
{
    __shared__ unsigned short As[2][BM * BK];
    __shared__ unsigned short Bs[2][BN * BK];

    const int tid  = threadIdx.x;
    const int lane = tid & 31;
    const int wave = tid >> 5;
    const int half = lane >> 4;
    const int r    = lane & 15;
    const int wm   = wave >> 1;          // 0..3: 16-row strip
    const int wn   = (wave & 1) * 2;     // 0/2:  pair of 16-col tiles

    const int tmB = blockIdx.y * BM;
    const int tnB = blockIdx.x * BN;

    v8f acc0 = {};
    v8f acc1 = {};
    const int KT = K / BK;

#if HAS_TDM
    if (tid < 32) {
        tdm_load_tile(A + (size_t)tmB * K, lds_off(&As[0][0]), BM, BK, K, K, Mp);
        tdm_load_tile(B + (size_t)tnB * K, lds_off(&Bs[0][0]), BN, BK, K, K, Np);
    }
#endif
    for (int kt = 0; kt < KT; ++kt) {
        const int buf = kt & 1;
#if HAS_TDM
        if (tid < 32) __builtin_amdgcn_s_wait_tensorcnt((short)0);
        __syncthreads();                  // tile kt resident in LDS buf
        if (kt + 1 < KT && tid < 32) {    // overlap: DMA tile kt+1 into other buf
            const int nb = (kt + 1) & 1;
            tdm_load_tile(A + (size_t)tmB * K + (size_t)(kt + 1) * BK,
                          lds_off(&As[nb][0]), BM, BK, K, K, Mp);
            tdm_load_tile(B + (size_t)tnB * K + (size_t)(kt + 1) * BK,
                          lds_off(&Bs[nb][0]), BN, BK, K, K, Np);
        }
#else
        {   // cooperative staging: 64x64 bf16 per matrix, 2x b128 per thread
            const int rowA = tid >> 2;
            const int colA = (tid & 3) * 16;
            const unsigned short* ga = A + (size_t)(tmB + rowA) * K + (size_t)kt * BK + colA;
            const unsigned short* gb = B + (size_t)(tnB + rowA) * K + (size_t)kt * BK + colA;
            __builtin_prefetch((const void*)(ga + BK), 0, 0);
            __builtin_prefetch((const void*)(gb + BK), 0, 0);
            *(uint4*)&As[buf][rowA * BK + colA]     = *(const uint4*)(ga);
            *(uint4*)&As[buf][rowA * BK + colA + 8] = *(const uint4*)(ga + 8);
            *(uint4*)&Bs[buf][rowA * BK + colA]     = *(const uint4*)(gb);
            *(uint4*)&Bs[buf][rowA * BK + colA + 8] = *(const uint4*)(gb + 8);
        }
        __syncthreads();
#endif
        const unsigned short* a0 = &As[buf][(wm * 16 + r) * BK];
        const unsigned short* b0 = &Bs[buf][(wn * 16 + r) * BK];
        const unsigned short* b1 = &Bs[buf][((wn + 1) * 16 + r) * BK];
#pragma unroll
        for (int kk = 0; kk < BK; kk += 32) {
            FragBF a, p, q;
            a.q[0] = *(const uint4*)(a0 + kk + half * 8);
            a.q[1] = *(const uint4*)(a0 + kk + 16 + half * 8);
            p.q[0] = *(const uint4*)(b0 + kk + half * 8);
            p.q[1] = *(const uint4*)(b0 + kk + 16 + half * 8);
            q.q[0] = *(const uint4*)(b1 + kk + half * 8);
            q.q[1] = *(const uint4*)(b1 + kk + 16 + half * 8);
            acc0 = __builtin_amdgcn_wmma_f32_16x16x32_bf16(
                false, a.v, false, p.v, (short)0, acc0, false, false);
            acc1 = __builtin_amdgcn_wmma_f32_16x16x32_bf16(
                false, a.v, false, q.v, (short)0, acc1, false, false);
        }
        __syncthreads();                  // done reading buf before it is refilled
    }

    const int row0 = tmB + wm * 16 + half * 8;
    float* cp0 = C + (size_t)row0 * Np + tnB + wn * 16 + r;
    float* cp1 = cp0 + 16;
#pragma unroll
    for (int v = 0; v < 8; ++v) {
        cp0[(size_t)v * Np] = acc0[v];
        cp1[(size_t)v * Np] = acc1[v];
    }
}

// ---------------- small utility kernels ----------------
__global__ void zero_f32(float* p, size_t n) {
    for (size_t i = (size_t)blockIdx.x * blockDim.x + threadIdx.x; i < n;
         i += (size_t)gridDim.x * blockDim.x) p[i] = 0.0f;
}

__global__ void build_idx(const int* __restrict__ s_idxs, int g, int L, int N,
                          int* __restrict__ idxval)
{
    int r = blockIdx.x * blockDim.x + threadIdx.x;
    if (r >= N) return;
    int lgsn = L / 10 + 1; if (lgsn > L / 2) lgsn = L / 2;
    int halfL = L / 2;
    int p;
    if (r < L) p = g * L + r;
    else { int e = r - L; p = (e < lgsn) ? e : (halfL + (e - lgsn)); }  // from group 0
    idxval[r] = s_idxs[p];
}

__global__ void build_idx_center(const int* __restrict__ s_idxs, int nb,
                                 int* __restrict__ idxc)
{
    int r = blockIdx.x * blockDim.x + threadIdx.x;
    if (r < nb) idxc[r] = s_idxs[r];   // s_idxs[0][:nb]
}

__global__ __launch_bounds__(256) void gather_rows_bf16(
    const float* __restrict__ src, const int* __restrict__ idx,
    int N, int Npad, int D,
    unsigned short* __restrict__ dst, float* __restrict__ sqnorm, int do_norm)
{
    __shared__ float red[256];
    int row = blockIdx.x;
    if (row >= Npad) return;
    if (row >= N) {
        for (int d = threadIdx.x; d < D; d += 256) dst[(size_t)row * D + d] = 0;
        if (threadIdx.x == 0 && sqnorm) sqnorm[row] = 0.0f;
        return;
    }
    int s = idx ? idx[row] : row;
    const float* x = src + (size_t)s * D;
    float acc = 0.0f;
    for (int d = threadIdx.x; d < D; d += 256) { float v = x[d]; acc += v * v; }
    red[threadIdx.x] = acc; __syncthreads();
    for (int o = 128; o > 0; o >>= 1) {
        if (threadIdx.x < o) red[threadIdx.x] += red[threadIdx.x + o];
        __syncthreads();
    }
    float ss = red[0];
    float inv = do_norm ? 1.0f / fmaxf(sqrtf(ss), 1e-12f) : 1.0f;
    for (int d = threadIdx.x; d < D; d += 256)
        dst[(size_t)row * D + d] = f2bf(x[d] * inv);
    if (threadIdx.x == 0 && sqnorm) sqnorm[row] = ss;
}

// W_P = exp(-max(2-2g, 0))  (SIGMA=1; unit-normalized rows) — in place.
__global__ void gram_to_wp(float* __restrict__ G, size_t n) {
    for (size_t i = (size_t)blockIdx.x * blockDim.x + threadIdx.x; i < n;
         i += (size_t)gridDim.x * blockDim.x) {
        float d2 = fmaxf(2.0f - 2.0f * G[i], 0.0f);
        G[i] = __expf(-d2);
    }
}

__global__ __launch_bounds__(256) void sdist_rowmean(
    const float* __restrict__ G, int N, int Npad, float* __restrict__ rm)
{
    __shared__ float red[256];
    int i = blockIdx.x; if (i >= N) return;
    const float* g = G + (size_t)i * Npad;
    float acc = 0.0f;
    for (int j = threadIdx.x; j < N; j += 256)
        acc += sqrtf(fmaxf(2.0f - 2.0f * g[j], 0.0f));
    red[threadIdx.x] = acc; __syncthreads();
    for (int o = 128; o > 0; o >>= 1) {
        if (threadIdx.x < o) red[threadIdx.x] += red[threadIdx.x + o];
        __syncthreads();
    }
    if (threadIdx.x == 0) rm[i] = fmaxf(red[0] / (float)N, 1e-20f);
}

#define NMAXP 1280
__global__ __launch_bounds__(256) void topk20(
    const float* __restrict__ WP, const int* __restrict__ idxval,
    int N, int Npad, int* __restrict__ tk)
{
    __shared__ float vals[NMAXP];
    __shared__ float rv[256];
    __shared__ int   ri[256];
    int i = blockIdx.x; if (i >= N) return;
    int myidx = idxval[i];
    for (int j = threadIdx.x; j < Npad; j += 256)
        vals[j] = (j < N) ? ((idxval[j] == myidx) ? 1.0f : WP[(size_t)i * Npad + j])
                          : -1e30f;
    __syncthreads();
    for (int t = 0; t < 20; ++t) {
        float bv = -1e30f; int bi = Npad;
        for (int j = threadIdx.x; j < N; j += 256) {
            float v = vals[j];
            if (v > bv || (v == bv && j < bi)) { bv = v; bi = j; }
        }
        rv[threadIdx.x] = bv; ri[threadIdx.x] = bi; __syncthreads();
        for (int o = 128; o > 0; o >>= 1) {
            if (threadIdx.x < o) {
                float ov = rv[threadIdx.x + o]; int oi = ri[threadIdx.x + o];
                if (ov > rv[threadIdx.x] || (ov == rv[threadIdx.x] && oi < ri[threadIdx.x])) {
                    rv[threadIdx.x] = ov; ri[threadIdx.x] = oi;
                }
            }
            __syncthreads();
        }
        if (threadIdx.x == 0) { tk[i * 20 + t] = ri[0]; vals[ri[0]] = -1e30f; }
        __syncthreads();
    }
}

__global__ void scatter_wnn(const int* __restrict__ tk, int N, int Npad,
                            float* __restrict__ WNN)
{
    int t = blockIdx.x * blockDim.x + threadIdx.x;
    if (t >= N * 20) return;
    int i = t / 20;
    WNN[(size_t)i * Npad + tk[t]] = 1.0f;
}

__global__ void build_v(const float* __restrict__ WNN, int N, int Npad,
                        float* __restrict__ V, unsigned short* __restrict__ Vb)
{
    size_t total = (size_t)Npad * Npad;
    for (size_t t = (size_t)blockIdx.x * blockDim.x + threadIdx.x; t < total;
         t += (size_t)gridDim.x * blockDim.x) {
        int i = (int)(t / Npad), j = (int)(t % Npad);
        float v = 0.0f;
        if (i < N && j < N &&
            WNN[(size_t)i * Npad + j] > 0.5f && WNN[(size_t)j * Npad + i] > 0.5f)
            v = 1.0f;
        V[t] = v;
        Vb[t] = f2bf(v);
    }
}

__global__ __launch_bounds__(256) void row_sum(
    const float* __restrict__ Mtx, int N, int Npad, float* __restrict__ out)
{
    __shared__ float red[256];
    int i = blockIdx.x; if (i >= N) return;
    const float* m = Mtx + (size_t)i * Npad;
    float acc = 0.0f;
    for (int j = threadIdx.x; j < N; j += 256) acc += m[j];
    red[threadIdx.x] = acc; __syncthreads();
    for (int o = 128; o > 0; o >>= 1) {
        if (threadIdx.x < o) red[threadIdx.x] += red[threadIdx.x + o];
        __syncthreads();
    }
    if (threadIdx.x == 0) out[i] = red[0];
}

__global__ void wc_tilda(float* __restrict__ VV, const float* __restrict__ V,
                         const float* __restrict__ cnt, int N, int Npad)
{
    size_t total = (size_t)Npad * Npad;
    for (size_t t = (size_t)blockIdx.x * blockDim.x + threadIdx.x; t < total;
         t += (size_t)gridDim.x * blockDim.x) {
        int i = (int)(t / Npad), j = (int)(t % Npad);
        float r = 0.0f;
        if (i < N && j < N) {
            float c = cnt[i];
            if (c > 0.0f) r = V[t] * VV[t] / fmaxf(c, 1.0f);
        }
        VV[t] = r;
    }
}

__global__ void wc_hat(const float* __restrict__ WCT, const int* __restrict__ tk,
                       int N, int Npad, float* __restrict__ WCH)
{
    size_t total = (size_t)N * N;
    for (size_t t = (size_t)blockIdx.x * blockDim.x + threadIdx.x; t < total;
         t += (size_t)gridDim.x * blockDim.x) {
        int i = (int)(t / N), j = (int)(t % N);
        float a = 0.0f;
#pragma unroll
        for (int h = 0; h < 10; ++h)
            a += WCT[(size_t)tk[i * 20 + h] * Npad + j];
        WCH[(size_t)i * Npad + j] = a * 0.1f;
    }
}

__global__ __launch_bounds__(256) void rc_accum(
    const float* __restrict__ G, const float* __restrict__ rm,
    const float* __restrict__ WP, const float* __restrict__ WCH,
    int N, int Npad, float scale, float* __restrict__ out)
{
    __shared__ float red[256];
    size_t total = (size_t)N * N;
    float acc = 0.0f;
    for (size_t t = (size_t)blockIdx.x * 256 + threadIdx.x; t < total;
         t += (size_t)gridDim.x * 256) {
        int i = (int)(t / N), j = (int)(t % N);
        if (i == j) continue;
        float wc = 0.5f * (WCH[(size_t)i * Npad + j] + WCH[(size_t)j * Npad + i]);
        float w  = 0.5f * (WP[(size_t)i * Npad + j] + wc);
        float sd = sqrtf(fmaxf(2.0f - 2.0f * G[(size_t)i * Npad + j], 0.0f)) / rm[i];
        float m  = fmaxf(1.0f - sd, 0.0f);           // DELTA = 1
        acc += sd * sd * w + m * m * (1.0f - w);
    }
    red[threadIdx.x] = acc; __syncthreads();
    for (int o = 128; o > 0; o >>= 1) {
        if (threadIdx.x < o) red[threadIdx.x] += red[threadIdx.x + o];
        __syncthreads();
    }
    if (threadIdx.x == 0) atomicAdd(out, red[0] * scale);
}

__global__ __launch_bounds__(256) void kl_stml_row(
    const float* __restrict__ Gf, const float* __restrict__ rmf,
    const float* __restrict__ Gg, const float* __restrict__ rmg,
    int N, int Npad, float scale, float* __restrict__ out)
{
    __shared__ float red[256];
    int i = blockIdx.x; if (i >= N) return;
    const float* gf = Gf + (size_t)i * Npad;
    const float* gg = Gg + (size_t)i * Npad;
    float imf = 1.0f / rmf[i], img = 1.0f / rmg[i];

    float mA = -1e30f, mB = -1e30f;
    for (int j = threadIdx.x; j < N; j += 256) {
        float A = -sqrtf(fmaxf(2.0f - 2.0f * gf[j], 0.0f)) * imf;
        float B = -sqrtf(fmaxf(2.0f - 2.0f * gg[j], 0.0f)) * img;
        mA = fmaxf(mA, A); mB = fmaxf(mB, B);
    }
    red[threadIdx.x] = mA; __syncthreads();
    for (int o = 128; o > 0; o >>= 1) { if (threadIdx.x < o) red[threadIdx.x] = fmaxf(red[threadIdx.x], red[threadIdx.x + o]); __syncthreads(); }
    mA = red[0]; __syncthreads();
    red[threadIdx.x] = mB; __syncthreads();
    for (int o = 128; o > 0; o >>= 1) { if (threadIdx.x < o) red[threadIdx.x] = fmaxf(red[threadIdx.x], red[threadIdx.x + o]); __syncthreads(); }
    mB = red[0]; __syncthreads();

    float sA = 0.0f, sB = 0.0f;
    for (int j = threadIdx.x; j < N; j += 256) {
        float A = -sqrtf(fmaxf(2.0f - 2.0f * gf[j], 0.0f)) * imf;
        float B = -sqrtf(fmaxf(2.0f - 2.0f * gg[j], 0.0f)) * img;
        sA += __expf(A - mA); sB += __expf(B - mB);
    }
    red[threadIdx.x] = sA; __syncthreads();
    for (int o = 128; o > 0; o >>= 1) { if (threadIdx.x < o) red[threadIdx.x] += red[threadIdx.x + o]; __syncthreads(); }
    sA = red[0]; __syncthreads();
    red[threadIdx.x] = sB; __syncthreads();
    for (int o = 128; o > 0; o >>= 1) { if (threadIdx.x < o) red[threadIdx.x] += red[threadIdx.x + o]; __syncthreads(); }
    sB = red[0]; __syncthreads();

    float lA = __logf(sA), lB = __logf(sB);
    float acc = 0.0f;
    for (int j = threadIdx.x; j < N; j += 256) {
        float A = -sqrtf(fmaxf(2.0f - 2.0f * gf[j], 0.0f)) * imf;
        float B = -sqrtf(fmaxf(2.0f - 2.0f * gg[j], 0.0f)) * img;
        float logq = A - mA - lA;
        float logp = B - mB - lB;
        float p = __expf(logp);
        acc += p * (logp - logq);
    }
    red[threadIdx.x] = acc; __syncthreads();
    for (int o = 128; o > 0; o >>= 1) { if (threadIdx.x < o) red[threadIdx.x] += red[threadIdx.x + o]; __syncthreads(); }
    if (threadIdx.x == 0) atomicAdd(out, red[0] * scale);
}

__global__ void make_logits(float* __restrict__ Cm, const float* __restrict__ sqx,
                            const float* __restrict__ sqc, int Nr, int Nc, int ld)
{
    size_t total = (size_t)Nr * Nc;
    for (size_t t = (size_t)blockIdx.x * blockDim.x + threadIdx.x; t < total;
         t += (size_t)gridDim.x * blockDim.x) {
        int i = (int)(t / Nc), j = (int)(t % Nc);
        float d2 = sqx[i] - 2.0f * Cm[(size_t)i * ld + j] + sqc[j];
        Cm[(size_t)i * ld + j] = sqrtf(fmaxf(d2, 0.0f));
    }
}

__global__ __launch_bounds__(256) void kl_center_row(
    const float* __restrict__ LP, const float* __restrict__ LQ,
    int Nr, int Nc, int ld, float scale, float* __restrict__ out)
{
    __shared__ float red[256];
    int i = blockIdx.x; if (i >= Nr) return;
    const float* lp = LP + (size_t)i * ld;
    const float* lq = LQ + (size_t)i * ld;
    const float eps = 1e-8f;

    float mP = -1e30f, mQ = -1e30f;
    for (int j = threadIdx.x; j < Nc; j += 256) { mP = fmaxf(mP, lp[j]); mQ = fmaxf(mQ, lq[j]); }
    red[threadIdx.x] = mP; __syncthreads();
    for (int o = 128; o > 0; o >>= 1) { if (threadIdx.x < o) red[threadIdx.x] = fmaxf(red[threadIdx.x], red[threadIdx.x + o]); __syncthreads(); }
    mP = red[0]; __syncthreads();
    red[threadIdx.x] = mQ; __syncthreads();
    for (int o = 128; o > 0; o >>= 1) { if (threadIdx.x < o) red[threadIdx.x] = fmaxf(red[threadIdx.x], red[threadIdx.x + o]); __syncthreads(); }
    mQ = red[0]; __syncthreads();

    float sP = 0.0f, sQ = 0.0f;
    for (int j = threadIdx.x; j < Nc; j += 256) { sP += __expf(lp[j] - mP); sQ += __expf(lq[j] - mQ); }
    red[threadIdx.x] = sP; __syncthreads();
    for (int o = 128; o > 0; o >>= 1) { if (threadIdx.x < o) red[threadIdx.x] += red[threadIdx.x + o]; __syncthreads(); }
    sP = red[0]; __syncthreads();
    red[threadIdx.x] = sQ; __syncthreads();
    for (int o = 128; o > 0; o >>= 1) { if (threadIdx.x < o) red[threadIdx.x] += red[threadIdx.x + o]; __syncthreads(); }
    sQ = red[0]; __syncthreads();

    float acc = 0.0f;
    for (int j = threadIdx.x; j < Nc; j += 256) {
        float p = __expf(lp[j] - mP) / sP;
        float q = __expf(lq[j] - mQ) / sQ;
        float pt = (p + eps) / (1.0f + eps);
        float qt = (q + eps) / (1.0f + eps);
        acc += pt * __logf(pt / qt);
    }
    red[threadIdx.x] = acc; __syncthreads();
    for (int o = 128; o > 0; o >>= 1) { if (threadIdx.x < o) red[threadIdx.x] += red[threadIdx.x + o]; __syncthreads(); }
    if (threadIdx.x == 0) atomicAdd(out, red[0] * scale);
}

// ---------------------------------------------------------------------------
extern "C" void kernel_launch(void* const* d_in, const int* in_sizes, int n_in,
                              void* d_out, int out_size, void* d_ws, size_t ws_size,
                              hipStream_t stream)
{
    (void)in_sizes; (void)n_in; (void)out_size; (void)ws_size;
    const int M = 8192, D = 512, C = 400, L = 1024;
    const int NMAX = NMAXP;          // 1280 = max N padded to 64
    const int CMR = 128, CMC = 448;  // padded center-logits GEMM dims

    const float* s_f    = (const float*)d_in[0];
    const float* s_g    = (const float*)d_in[1];
    const float* t_g    = (const float*)d_in[2];
    const float* center = (const float*)d_in[3];
    const int*   s_idxs = (const int*)d_in[4];
    float* out = (float*)d_out;

    size_t off = 0;
    char* base = (char*)d_ws;
    auto alloc = [&](size_t bytes) -> void* {
        off = (off + 255) & ~(size_t)255;
        void* p = base + off; off += bytes; return p;
    };
    unsigned short* SFB = (unsigned short*)alloc((size_t)NMAX * D * 2);
    unsigned short* SGB = (unsigned short*)alloc((size_t)NMAX * D * 2);
    unsigned short* TB  = (unsigned short*)alloc((size_t)NMAX * D * 2);
    unsigned short* CB  = (unsigned short*)alloc((size_t)CMC * D * 2);
    unsigned short* XB0 = (unsigned short*)alloc((size_t)CMR * D * 2);
    unsigned short* XB1 = (unsigned short*)alloc((size_t)CMR * D * 2);
    unsigned short* VB  = (unsigned short*)alloc((size_t)NMAX * NMAX * 2);
    float* GF  = (float*)alloc((size_t)NMAX * NMAX * 4);
    float* GG  = (float*)alloc((size_t)NMAX * NMAX * 4);
    float* WP  = (float*)alloc((size_t)NMAX * NMAX * 4);
    float* WNN = (float*)alloc((size_t)NMAX * NMAX * 4);
    float* V   = (float*)alloc((size_t)NMAX * NMAX * 4);
    float* VV  = (float*)alloc((size_t)NMAX * NMAX * 4);
    float* WCH = (float*)alloc((size_t)NMAX * NMAX * 4);
    int*   IDX = (int*)alloc((size_t)NMAX * 4);
    int*   IDXC= (int*)alloc((size_t)CMR * 4);
    int*   TK  = (int*)alloc((size_t)NMAX * 20 * 4);
    float* CNT = (float*)alloc((size_t)NMAX * 4);
    float* RMF = (float*)alloc((size_t)NMAX * 4);
    float* RMG = (float*)alloc((size_t)NMAX * 4);
    float* SQC = (float*)alloc((size_t)CMC * 4);
    float* SQX0= (float*)alloc((size_t)CMR * 4);
    float* SQX1= (float*)alloc((size_t)CMR * 4);
    float* CM0 = (float*)alloc((size_t)CMR * CMC * 4);
    float* CM1 = (float*)alloc((size_t)CMR * CMC * 4);

    zero_f32<<<1, 64, 0, stream>>>(out, 1);

    // center -> bf16 (padded to 448 rows) + squared norms
    gather_rows_bf16<<<CMC, 256, 0, stream>>>(center, nullptr, C, CMC, D, CB, SQC, 0);

    const int lgsn = L / 10 + 1;   // 103
    for (int g = 0; g < 2; ++g) {
        const int N    = (g == 0) ? L : (L + 2 * lgsn);       // 1024 / 1230
        const int Npad = (N + 63) & ~63;                       // 1024 / 1280
        const size_t NN = (size_t)Npad * Npad;
        const float* sf = s_f + (size_t)g * M * D;
        const float* sg = s_g + (size_t)g * M * D;
        const float* tg = t_g + (size_t)g * M * D;

        build_idx<<<(N + 255) / 256, 256, 0, stream>>>(s_idxs, g, L, N, IDX);

        gather_rows_bf16<<<Npad, 256, 0, stream>>>(sf, IDX, N, Npad, D, SFB, nullptr, 1);
        gather_rows_bf16<<<Npad, 256, 0, stream>>>(sg, IDX, N, Npad, D, SGB, nullptr, 1);
        gather_rows_bf16<<<Npad, 256, 0, stream>>>(tg, IDX, N, Npad, D, TB,  nullptr, 1);

        dim3 gtile(Npad / BN, Npad / BM);
        wmma_gemm_nt_tiled<<<gtile, 256, 0, stream>>>(SFB, SFB, GF, Npad, Npad, D);
        wmma_gemm_nt_tiled<<<gtile, 256, 0, stream>>>(SGB, SGB, GG, Npad, Npad, D);
        wmma_gemm_nt_tiled<<<gtile, 256, 0, stream>>>(TB,  TB,  WP, Npad, Npad, D);

        gram_to_wp<<<1024, 256, 0, stream>>>(WP, NN);
        topk20<<<N, 256, 0, stream>>>(WP, IDX, N, Npad, TK);
        zero_f32<<<1024, 256, 0, stream>>>(WNN, NN);
        scatter_wnn<<<(N * 20 + 255) / 256, 256, 0, stream>>>(TK, N, Npad, WNN);
        build_v<<<1024, 256, 0, stream>>>(WNN, N, Npad, V, VB);
        row_sum<<<N, 256, 0, stream>>>(V, N, Npad, CNT);

        // VV = V @ V  (V symmetric => V @ V^T on the WMMA path)
        wmma_gemm_nt_tiled<<<gtile, 256, 0, stream>>>(VB, VB, VV, Npad, Npad, Npad);
        wc_tilda<<<1024, 256, 0, stream>>>(VV, V, CNT, N, Npad);
        wc_hat<<<1024, 256, 0, stream>>>(VV, TK, N, Npad, WCH);

        sdist_rowmean<<<N, 256, 0, stream>>>(GF, N, Npad, RMF);
        sdist_rowmean<<<N, 256, 0, stream>>>(GG, N, Npad, RMG);

        float rc_scale = 1.0f / (8.0f * (float)N * (float)(N - 1));   // (/2)(/4)
        rc_accum<<<1024, 256, 0, stream>>>(GF, RMF, WP, WCH, N, Npad, rc_scale, out);
        rc_accum<<<1024, 256, 0, stream>>>(GG, RMG, WP, WCH, N, Npad, rc_scale, out);

        float kl_scale = 1.0f / (4.0f * (float)N);
        kl_stml_row<<<N, 256, 0, stream>>>(GF, RMF, GG, RMG, N, Npad, kl_scale, out);
    }

    // cross-group center KL (i=1, j=0): pidx = s_idxs[0][:103]
    build_idx_center<<<1, 128, 0, stream>>>(s_idxs, lgsn, IDXC);
    gather_rows_bf16<<<CMR, 256, 0, stream>>>(s_f,                 IDXC, lgsn, CMR, D, XB0, SQX0, 0);
    gather_rows_bf16<<<CMR, 256, 0, stream>>>(s_f + (size_t)M * D, IDXC, lgsn, CMR, D, XB1, SQX1, 0);
    dim3 gc(CMC / BN, CMR / BM);
    wmma_gemm_nt_tiled<<<gc, 256, 0, stream>>>(XB0, CB, CM0, CMR, CMC, D);
    wmma_gemm_nt_tiled<<<gc, 256, 0, stream>>>(XB1, CB, CM1, CMR, CMC, D);
    make_logits<<<256, 256, 0, stream>>>(CM0, SQX0, SQC, lgsn, C, CMC);
    make_logits<<<256, 256, 0, stream>>>(CM1, SQX1, SQC, lgsn, C, CMC);
    kl_center_row<<<lgsn, 256, 0, stream>>>(CM0, CM1, lgsn, C, CMC, 1.0f / (float)lgsn, out);
}
